// BertSelfAttention_50852412785290
// MI455X (gfx1250) — compile-verified, matrix operand order checked by reading, not run on hardware
//
#include <hip/hip_runtime.h>

// Problem constants (match reference)
#define B_   32
#define S_   256
#define D_   768
#define H_   12
#define M_   8
#define E_   64
#define HD_  64
#define BS_  (B_ * S_)     // 8192 rows
#define M1_  (M_ + 1)      // 9
#define WSZ_ ((size_t)D_ * D_)   // 589824 weight elems

typedef __attribute__((ext_vector_type(16))) __bf16 bf16x16;
typedef __attribute__((ext_vector_type(8)))  __bf16 bf16x8;
typedef __attribute__((ext_vector_type(8)))  float  f32x8;

// gcc-style int4 with explicit address spaces, matching the async builtin's
// parameter types ('int __vector(4) __device__ *' / '... __shared__ *').
typedef int v4i __attribute__((vector_size(16)));
typedef v4i __attribute__((address_space(1))) ga_v4i;   // global
typedef v4i __attribute__((address_space(3))) ls_v4i;   // LDS

#define TK   32
#define BSTR 40   // padded LDS row stride in bf16 elems (80B staggers banks)

// CDNA5 async global->LDS path (ASYNCcnt-tracked), with portable fallback.
#if defined(__has_builtin)
#if __has_builtin(__builtin_amdgcn_global_load_async_to_lds_b128) && \
    __has_builtin(__builtin_amdgcn_s_wait_asynccnt)
#define USE_ASYNC_LDS 1
#endif
#endif
#ifndef USE_ASYNC_LDS
#define USE_ASYNC_LDS 0
#endif

// ---------------------------------------------------------------------------
// One-time f32 -> bf16 conversion (row-major), vectorized 4 elems/thread.
// ---------------------------------------------------------------------------
__global__ __launch_bounds__(256) void cvt_bf16(const float* __restrict__ src,
                                                __bf16* __restrict__ dst, int n4)
{
    int i = blockIdx.x * 256 + threadIdx.x;
    if (i >= n4) return;
    float4 f = reinterpret_cast<const float4*>(src)[i];
    union { __bf16 h[4]; uint2 u; } pk;
    pk.h[0] = (__bf16)f.x; pk.h[1] = (__bf16)f.y;
    pk.h[2] = (__bf16)f.z; pk.h[3] = (__bf16)f.w;
    reinterpret_cast<uint2*>(dst)[i] = pk.u;
}

// ---------------------------------------------------------------------------
// One-time weight convert+transpose: T[n][k] = (bf16)W[k][n]  (768x768).
// blockIdx.z selects Wq/Wk/Wv. One-shot cost, ~3.4 MB total output.
// ---------------------------------------------------------------------------
__global__ __launch_bounds__(256) void cvt_wT(
    const float* __restrict__ W0, const float* __restrict__ W1,
    const float* __restrict__ W2,
    __bf16* __restrict__ T0, __bf16* __restrict__ T1, __bf16* __restrict__ T2)
{
    const float* W = (blockIdx.z == 0) ? W0 : (blockIdx.z == 1) ? W1 : W2;
    __bf16*      T = (blockIdx.z == 0) ? T0 : (blockIdx.z == 1) ? T1 : T2;
    int idx = blockIdx.x * 256 + threadIdx.x;       // 0 .. 589823
    int k = idx % D_;                               // consecutive tid -> consecutive k
    int n = idx / D_;
    T[(size_t)n * D_ + k] = (__bf16)W[(size_t)k * D_ + n];
}

// ---------------------------------------------------------------------------
// Fused Q/K/V projection GEMM: C = X[8192x768] * W[768x768] + bias.
// blockIdx.z selects the weight set. 8 wave32s per block, 128x128 block tile,
// 2x4 v_wmma_f32_16x16x32_bf16 tiles per wave. A-fragments come straight from
// global bf16 (two b128 per tile, ISA 16-bit A layout). B tile (from the
// pre-transposed bf16 weight) is double-buffered in LDS, staged with
// global_load_async_to_lds_b128 when available.
// ---------------------------------------------------------------------------
__global__ __launch_bounds__(256) void qkv_gemm(
    const __bf16* __restrict__ Xb,
    const __bf16* __restrict__ Wt0, const __bf16* __restrict__ Wt1,
    const __bf16* __restrict__ Wt2,
    const float* __restrict__ bq, const float* __restrict__ bk,
    const float* __restrict__ bv,
    float* __restrict__ Qo, float* __restrict__ Ko, float* __restrict__ Vo)
{
    const __bf16* Wt; const float* bias; float* C;
    if (blockIdx.z == 0)      { Wt = Wt0; bias = bq; C = Qo; }
    else if (blockIdx.z == 1) { Wt = Wt1; bias = bk; C = Ko; }
    else                      { Wt = Wt2; bias = bv; C = Vo; }

    __shared__ __align__(16) __bf16 Bs[2][128 * BSTR];   // 2 x 10 KB

    const int tid  = threadIdx.x;
    const int lane = tid & 31;
    const int w    = tid >> 5;
    const int wr   = w & 3;               // 4 row groups of 32
    const int wc   = w >> 2;              // 2 col groups of 64
    const int lr   = lane & 15;
    const int kg   = (lane >> 4) * 8;     // K-group base (ISA 16-bit layout)
    const int rowBase = blockIdx.y * 128;
    const int colBase = blockIdx.x * 128;

    // stage one 128x32 bf16 B tile (stored [col][k]) into Bs[buf]
    auto stage = [&](int buf, int k0) {
#pragma unroll
        for (int j = 0; j < 2; ++j) {
            int i4  = tid + j * 256;            // 512 x 16B segments
            int col = i4 >> 2;
            int seg = i4 & 3;
            const __bf16* src = Wt + (size_t)(colBase + col) * D_ + k0 + seg * 8;
            __bf16* dst = &Bs[buf][col * BSTR + seg * 8];
#if USE_ASYNC_LDS
            __builtin_amdgcn_global_load_async_to_lds_b128(
                (ga_v4i*)src, (ls_v4i*)dst, 0, 0);
#else
            *reinterpret_cast<uint4*>(dst) =
                *reinterpret_cast<const uint4*>(src);
#endif
        }
    };

    // per-lane A fragment base pointers (direct-from-global, L2 resident)
    const __bf16* Ap[2];
    Ap[0] = Xb + (size_t)(rowBase + wr * 32 +  0 + lr) * D_ + kg;
    Ap[1] = Xb + (size_t)(rowBase + wr * 32 + 16 + lr) * D_ + kg;

    f32x8 acc[2][4];
#pragma unroll
    for (int mt = 0; mt < 2; ++mt)
#pragma unroll
        for (int nt = 0; nt < 4; ++nt)
#pragma unroll
            for (int e = 0; e < 8; ++e) acc[mt][nt][e] = 0.0f;

    stage(0, 0);
#if USE_ASYNC_LDS
    __builtin_amdgcn_s_wait_asynccnt(0);
#endif
    __syncthreads();

    for (int ks = 0; ks < D_ / TK; ++ks) {
        const int k0  = ks * TK;
        const int cur = ks & 1;
        if (ks + 1 < D_ / TK) stage(cur ^ 1, k0 + TK);

        // fragments: v[0..7] = K kg..kg+7, v[8..15] = K 16+kg..16+kg+7
        bf16x16 af[2], bfr[4];
#pragma unroll
        for (int mt = 0; mt < 2; ++mt) {
            bf16x8 lo = *reinterpret_cast<const bf16x8*>(Ap[mt] + k0);
            bf16x8 hi = *reinterpret_cast<const bf16x8*>(Ap[mt] + k0 + 16);
#pragma unroll
            for (int i = 0; i < 8; ++i) { af[mt][i] = lo[i]; af[mt][i + 8] = hi[i]; }
        }
#pragma unroll
        for (int nt = 0; nt < 4; ++nt) {
            int base = (wc * 64 + nt * 16 + lr) * BSTR + kg;
            bf16x8 lo = *reinterpret_cast<const bf16x8*>(&Bs[cur][base]);
            bf16x8 hi = *reinterpret_cast<const bf16x8*>(&Bs[cur][base + 16]);
#pragma unroll
            for (int i = 0; i < 8; ++i) { bfr[nt][i] = lo[i]; bfr[nt][i + 8] = hi[i]; }
        }

#pragma unroll
        for (int mt = 0; mt < 2; ++mt)
#pragma unroll
            for (int nt = 0; nt < 4; ++nt)
                acc[mt][nt] = __builtin_amdgcn_wmma_f32_16x16x32_bf16(
                    false, af[mt], false, bfr[nt], (short)0, acc[mt][nt],
                    false, false);

#if USE_ASYNC_LDS
        __builtin_amdgcn_s_wait_asynccnt(0);
#endif
        __syncthreads();
    }

    // epilogue: D layout lane->N, vgpr(+8*hi-half)->M; add bias
    const int hi8 = (lane >> 4) * 8;
#pragma unroll
    for (int mt = 0; mt < 2; ++mt)
#pragma unroll
        for (int nt = 0; nt < 4; ++nt) {
            int gcol = colBase + wc * 64 + nt * 16 + lr;
            float bb = bias[gcol];
            int grow = rowBase + wr * 32 + mt * 16 + hi8;
#pragma unroll
            for (int e = 0; e < 8; ++e)
                C[(size_t)(grow + e) * D_ + gcol] = acc[mt][nt][e] + bb;
        }
}

// ---------------------------------------------------------------------------
// Pass B: 3-token n-gram attention. One thread per (b,s,h).
// Key/value rows are {s+1 (or 1 at s=S-1), s, 0} of the shared K/V GEMM output.
// ---------------------------------------------------------------------------
__global__ __launch_bounds__(256) void ngram_attn(
    const float* __restrict__ Q, const float* __restrict__ K,
    const float* __restrict__ V, float* __restrict__ ctx)
{
    int t  = blockIdx.x * 256 + threadIdx.x;     // 0..98303
    int h  = t % H_;
    int bs = t / H_;
    int s  = bs % S_;
    int b  = bs / S_;
    int sl = (s == S_ - 1) ? 1 : s + 1;

    size_t rq = (size_t)bs * D_ + h * HD_;                 // also n=1 row
    size_t r0 = (size_t)(b * S_ + sl) * D_ + h * HD_;      // n=0 (h_last)
    size_t r2 = (size_t)(b * S_) * D_ + h * HD_;           // n=2 (h[:,0])

    float d0 = 0.f, d1 = 0.f, d2 = 0.f;
#pragma unroll
    for (int i = 0; i < HD_; i += 4) {
        float4 q  = *reinterpret_cast<const float4*>(Q + rq + i);
        float4 k0 = *reinterpret_cast<const float4*>(K + r0 + i);
        float4 k1 = *reinterpret_cast<const float4*>(K + rq + i);
        float4 k2 = *reinterpret_cast<const float4*>(K + r2 + i);
        d0 += q.x * k0.x + q.y * k0.y + q.z * k0.z + q.w * k0.w;
        d1 += q.x * k1.x + q.y * k1.y + q.z * k1.z + q.w * k1.w;
        d2 += q.x * k2.x + q.y * k2.y + q.z * k2.z + q.w * k2.w;
    }
    const float sc = 0.125f;                     // 1/sqrt(64)
    d0 *= sc; d1 *= sc; d2 *= sc;
    float mx = fmaxf(d0, fmaxf(d1, d2));
    float e0 = __expf(d0 - mx), e1 = __expf(d1 - mx), e2 = __expf(d2 - mx);
    float inv = 1.f / (e0 + e1 + e2);
    e0 *= inv; e1 *= inv; e2 *= inv;

#pragma unroll
    for (int i = 0; i < HD_; i += 4) {
        float4 v0 = *reinterpret_cast<const float4*>(V + r0 + i);
        float4 v1 = *reinterpret_cast<const float4*>(V + rq + i);
        float4 v2 = *reinterpret_cast<const float4*>(V + r2 + i);
        float4 o;
        o.x = e0 * v0.x + e1 * v1.x + e2 * v2.x;
        o.y = e0 * v0.y + e1 * v1.y + e2 * v2.y;
        o.z = e0 * v0.z + e1 * v1.z + e2 * v2.z;
        o.w = e0 * v0.w + e1 * v1.w + e2 * v2.w;
        *reinterpret_cast<float4*>(ctx + rq + i) = o;
    }
}

// ---------------------------------------------------------------------------
// Pass C: dependency-star attention, one thread per (b,s,h).
// Edge projections folded algebraically:
//   logit_m = (q.star_m + demb_m.(Wek_h @ q) + q.bek_h) * sc + mask_m
//   out     = sum_m p_m star_m + (sum_m p_m demb_m) @ Wev_h + bev_h
// (avoids the 2x226MB [B,S,9,768] intermediates entirely)
// ---------------------------------------------------------------------------
__global__ __launch_bounds__(128) void star_attn(
    const float* __restrict__ ctx, const float* __restrict__ hs,
    const float* __restrict__ demb, const int* __restrict__ adj,
    const float* __restrict__ mask,
    const float* __restrict__ Wek, const float* __restrict__ bek,
    const float* __restrict__ Wev, const float* __restrict__ bev,
    float* __restrict__ out)
{
    int t   = blockIdx.x * 128 + threadIdx.x;    // 0..98303
    int h   = t % H_;
    int bs  = t / H_;
    int b   = bs / S_;
    int off = h * HD_;

    float q[HD_];
    const float* qp = ctx + (size_t)bs * D_ + off;
#pragma unroll
    for (int d = 0; d < HD_; ++d) q[d] = qp[d];

    // GK = Wek_h @ q  (E=64 per-head projection of the query)
    float GK[E_];
#pragma unroll
    for (int e = 0; e < E_; ++e) {
        const float* wr = Wek + (size_t)e * D_ + off;
        float a = 0.f;
#pragma unroll
        for (int d = 0; d < HD_; ++d) a += wr[d] * q[d];
        GK[e] = a;
    }
    float qbk = 0.f;
#pragma unroll
    for (int d = 0; d < HD_; ++d) qbk += q[d] * bek[off + d];

    const float* db = demb + (size_t)bs * M1_ * E_;
    const int*   ix = adj  + (size_t)bs * M_;
    const float* mk = mask + (size_t)bs * M1_;

    float lg[M1_];
    const float sc = 0.125f;
#pragma unroll
    for (int m = 0; m < M1_; ++m) {
        const float* srow = (m < M_)
            ? ctx + (size_t)(b * S_ + ix[m]) * D_ + off
            : hs  + (size_t)bs * D_ + off;
        float a = 0.f;
#pragma unroll
        for (int d = 0; d < HD_; ++d) a += q[d] * srow[d];
        const float* dm = db + m * E_;
        float ea = 0.f;
#pragma unroll
        for (int e = 0; e < E_; ++e) ea += dm[e] * GK[e];
        lg[m] = (a + ea + qbk) * sc + mk[m];
    }

    float mx = lg[0];
#pragma unroll
    for (int m = 1; m < M1_; ++m) mx = fmaxf(mx, lg[m]);
    float p[M1_], sum = 0.f;
#pragma unroll
    for (int m = 0; m < M1_; ++m) { p[m] = __expf(lg[m] - mx); sum += p[m]; }
    float inv = 1.f / sum;
#pragma unroll
    for (int m = 0; m < M1_; ++m) p[m] *= inv;

    float pd[E_];
#pragma unroll
    for (int e = 0; e < E_; ++e) {
        float a = 0.f;
#pragma unroll
        for (int m = 0; m < M1_; ++m) a += p[m] * db[m * E_ + e];
        pd[e] = a;
    }

    float o[HD_];
#pragma unroll
    for (int d = 0; d < HD_; ++d) o[d] = 0.f;
#pragma unroll
    for (int m = 0; m < M1_; ++m) {
        const float* srow = (m < M_)
            ? ctx + (size_t)(b * S_ + ix[m]) * D_ + off
            : hs  + (size_t)bs * D_ + off;
        float pm = p[m];
#pragma unroll
        for (int d = 0; d < HD_; ++d) o[d] += pm * srow[d];
    }
#pragma unroll
    for (int e = 0; e < E_; ++e) {
        const float* wv = Wev + (size_t)e * D_ + off;
        float pe = pd[e];
#pragma unroll
        for (int d = 0; d < HD_; ++d) o[d] += pe * wv[d];
    }
    float* op = out + (size_t)bs * D_ + off;
#pragma unroll
    for (int d = 0; d < HD_; ++d) op[d] = o[d] + bev[off + d];
}

// ---------------------------------------------------------------------------
extern "C" void kernel_launch(void* const* d_in, const int* in_sizes, int n_in,
                              void* d_out, int out_size, void* d_ws, size_t ws_size,
                              hipStream_t stream) {
    (void)in_sizes; (void)n_in; (void)out_size; (void)ws_size;
    const float* hs   = (const float*)d_in[1];
    const float* demb = (const float*)d_in[2];
    const int*   adj  = (const int*)  d_in[3];
    const float* mask = (const float*)d_in[4];
    const float* Wq = (const float*)d_in[5];   const float* bq  = (const float*)d_in[6];
    const float* Wk = (const float*)d_in[7];   const float* bk  = (const float*)d_in[8];
    const float* Wv = (const float*)d_in[9];   const float* bv  = (const float*)d_in[10];
    const float* Wek = (const float*)d_in[11]; const float* bek = (const float*)d_in[12];
    const float* Wev = (const float*)d_in[13]; const float* bev = (const float*)d_in[14];
    float* out = (float*)d_out;

    // workspace: Q | K | V | CTX (f32, 24 MiB each) | Xb (bf16) | 3x WtT (bf16)
    const size_t mat = (size_t)BS_ * D_;       // 6291456
    float*  Qb  = (float*)d_ws;
    float*  Kb  = Qb + mat;
    float*  Vb  = Kb + mat;
    float*  CT  = Vb + mat;
    __bf16* Xb  = (__bf16*)(CT + mat);
    __bf16* Wt0 = Xb + mat;
    __bf16* Wt1 = Wt0 + WSZ_;
    __bf16* Wt2 = Wt1 + WSZ_;

    // 0) one-time bf16 conversion / weight transpose
    cvt_bf16<<<dim3((int)(mat / 4 / 256)), 256, 0, stream>>>(hs, Xb, (int)(mat / 4));
    cvt_wT<<<dim3((int)(WSZ_ / 256), 1, 3), 256, 0, stream>>>(Wq, Wk, Wv, Wt0, Wt1, Wt2);

    // 1) fused Q/K/V projections on the WMMA pipe
    qkv_gemm<<<dim3(D_ / 128, BS_ / 128, 3), 256, 0, stream>>>(
        Xb, Wt0, Wt1, Wt2, bq, bk, bv, Qb, Kb, Vb);

    // 2) 3-token n-gram attention -> ctx
    ngram_attn<<<dim3((BS_ * H_) / 256), 256, 0, stream>>>(Qb, Kb, Vb, CT);

    // 3) dependency-star attention with folded edge projections -> out
    star_attn<<<dim3((BS_ * H_) / 128), 128, 0, stream>>>(
        CT, hs, demb, adj, mask, Wek, bek, Wev, bev, out);
}
